// GaussianLSS_75677323755606
// MI455X (gfx1250) — matrix operand order repaired and checked by reading, not compiled, not run on hardware
//
#include <hip/hip_runtime.h>
#include <hip/hip_bf16.h>

typedef __attribute__((ext_vector_type(16))) _Float16 v16h;
typedef __attribute__((ext_vector_type(8)))  _Float16 v8h;
typedef __attribute__((ext_vector_type(8)))  float    v8f;

#define G_TOT 2048
#define D_CH  64

// broadcast a float from lane k (k uniform) -> v_readlane_b32
__device__ __forceinline__ float lane_bcast(float v, int k) {
  return __builtin_bit_cast(float, __builtin_amdgcn_readlane(__builtin_bit_cast(int, v), k));
}

// A-fragment K mapping for 16-bit 16x32 A matrices (contiguous k = e + 16*(lane/16), same order as B)
__device__ __forceinline__ int kmapA(int lane, int e) {
  return e + ((lane >> 4) << 4);
}

// ---------------------------------------------------------------------------
// Prep: per-gaussian 2D splat parameters for both stages
// prm layout per stage: [u | v | A | B | C | opEff] each G floats
// ---------------------------------------------------------------------------
__global__ __launch_bounds__(256) void glss_prep_params(
    const float* __restrict__ means, const float* __restrict__ cov6,
    const float* __restrict__ opac, float* __restrict__ prm0,
    float* __restrict__ prm1) {
  int g = blockIdx.x * 256 + threadIdx.x;
  if (g >= G_TOT) return;
  float mx = means[g * 3 + 0], my = means[g * 3 + 1], mz = means[g * 3 + 2];
  bool pos = (mx >= -50.f) && (mx <= 50.f) && (my >= -50.f) && (my <= 50.f) &&
             (mz >= -4.f) && (mz <= 4.f);
  float c00 = cov6[g * 6 + 0], c01 = cov6[g * 6 + 1], c11 = cov6[g * 6 + 3];
  #pragma unroll
  for (int s = 0; s < 2; ++s) {
    float H = s ? 200.f : 100.f;
    float sh = H / 100.f;                       // sh == sw (square)
    float u = -sh * my + 0.5f * H;
    float v = -sh * mx + 0.5f * H;
    float a = sh * sh * c11 + 0.3f;
    float c = sh * sh * c00 + 0.3f;
    float bb = sh * sh * c01;
    float det = fmaxf(a * c - bb * bb, 1e-8f);
    float* prm = s ? prm1 : prm0;
    float op = opac[g * 2 + s];
    bool mk = (op > 0.05f) && pos;
    prm[g]             = u;
    prm[G_TOT + g]     = v;
    prm[2 * G_TOT + g] = c / det;
    prm[3 * G_TOT + g] = -bb / det;
    prm[4 * G_TOT + g] = a / det;
    prm[5 * G_TOT + g] = mk ? op : 0.f;
  }
}

// ---------------------------------------------------------------------------
// Prep: swizzle features (G,D) f32 -> WMMA A-fragment layout f16
// featA[((chunk*4 + mtile)*32 + lane)*16 + e] = feat[g][d]
// ---------------------------------------------------------------------------
__global__ __launch_bounds__(256) void glss_prep_featA(
    const float* __restrict__ feat, _Float16* __restrict__ featA) {
  int idx = blockIdx.x * 256 + threadIdx.x;  // 64*4*32*16 = 131072
  if (idx >= (G_TOT / 32) * 4 * 32 * 16) return;
  int e = idx & 15, L = (idx >> 4) & 31, m = (idx >> 9) & 3, c = idx >> 11;
  int g = c * 32 + kmapA(L, e);
  int d = m * 16 + (L & 15);
  featA[idx] = (_Float16)feat[g * D_CH + d];
}

// ---------------------------------------------------------------------------
// Prep: swizzle all 11 conv weight tensors into A-fragment layout f16
// per conv: [tap(9)][kc(2)][mtile(4)][lane(32)][e(16)] = 36864 halves
// conv order: 0:c1[0] 1:c1[1] 2..9:blk[s][b][j] 10:up[0]
// ---------------------------------------------------------------------------
__global__ __launch_bounds__(256) void glss_prep_wfrag(
    const float* __restrict__ c1w, const float* __restrict__ blkw,
    const float* __restrict__ upw, _Float16* __restrict__ wf) {
  int idx = blockIdx.x * 256 + threadIdx.x;  // 11 * 36864 = 405504
  if (idx >= 11 * 36864) return;
  int conv = idx / 36864;
  int w = idx % 36864;
  int t = w >> 12;          // 0..8
  int rem = w & 4095;
  int kc = rem >> 11;
  int m = (rem >> 9) & 3;
  int L = (rem >> 4) & 31;
  int e = rem & 15;
  int o = m * 16 + (L & 15);
  int i = kc * 32 + kmapA(L, e);
  int r = t / 3, s = t % 3;
  const float* src;
  if (conv == 0)      src = c1w;
  else if (conv == 1) src = c1w + 64 * 64 * 9;
  else if (conv < 10) src = blkw + (size_t)(conv - 2) * 64 * 64 * 9;
  else                src = upw;
  wf[idx] = (_Float16)src[((o * 64 + i) * 3 + r) * 3 + s];
}

__global__ __launch_bounds__(256) void glss_prep_emb(
    const float* __restrict__ emb, _Float16* __restrict__ out) {
  int idx = blockIdx.x * 256 + threadIdx.x;  // 10000*64
  if (idx < 10000 * 64) out[idx] = (_Float16)emb[idx];
}

// ---------------------------------------------------------------------------
// Splatting render: wave owns 32 pixels; sequential front-to-back scan over
// gaussians computes w = alpha * T_exc into LDS (f16); every 32 gaussians,
// out[64,32px] += feat[64,32g] x w[32g,32px] via 8x v_wmma_f32_16x16x32_f16.
// Output: channel-last f16 [HW][64].
// ---------------------------------------------------------------------------
template <int HH>
__global__ __launch_bounds__(256) void glss_render(
    const float* __restrict__ prm, const _Float16* __restrict__ featA,
    _Float16* __restrict__ out) {
  constexpr int W = HH, HW = HH * HH;
  const int lane = threadIdx.x & 31;
  const int wid = threadIdx.x >> 5;
  const int p0 = (blockIdx.x * 8 + wid) * 32;
  const int hi = lane >> 4, lo = lane & 15;
  __shared__ __align__(32) _Float16 wlds[8][32][48];  // [wave][pixel][K] +pad
  if (p0 >= HW) return;

  const int p = p0 + lane;  // pixel owned by this lane for the scan
  const float px = (float)(p % W);
  const float py = (float)(p / W);

  v8f zf = {};
  v8f acc[4][2];
  #pragma unroll
  for (int m = 0; m < 4; ++m) { acc[m][0] = zf; acc[m][1] = zf; }

  const float* Us = prm;
  const float* Vs = prm + G_TOT;
  const float* As = prm + 2 * G_TOT;
  const float* Bs = prm + 3 * G_TOT;
  const float* Cs = prm + 4 * G_TOT;
  const float* Os = prm + 5 * G_TOT;

  float T = 1.0f;
  for (int c = 0; c < G_TOT / 32; ++c) {
    const int g0 = c * 32;
    const float lu = Us[g0 + lane], lv = Vs[g0 + lane], la = As[g0 + lane];
    const float lb = Bs[g0 + lane], lc = Cs[g0 + lane], lop = Os[g0 + lane];
    for (int k = 0; k < 32; ++k) {
      const float u = lane_bcast(lu, k);
      const float v = lane_bcast(lv, k);
      const float A = lane_bcast(la, k);
      const float Bc = lane_bcast(lb, k);
      const float C = lane_bcast(lc, k);
      const float op = lane_bcast(lop, k);
      const float dx = u - px, dy = v - py;
      const float power = -0.5f * (A * dx * dx + C * dy * dy) - Bc * dx * dy;
      float alpha = fminf(op * __expf(power), 0.99f);
      alpha = (alpha >= (1.0f / 255.0f) && power <= 0.0f) ? alpha : 0.0f;
      wlds[wid][lane][k] = (_Float16)(alpha * T);
      T *= (1.0f - alpha);
    }
    // B fragments: element e -> K = e + 16*hi, N = lo  (contiguous LDS 32B)
    v16h b0 = *(const v16h*)&wlds[wid][lo][hi * 16];
    v16h b1 = *(const v16h*)&wlds[wid][16 + lo][hi * 16];
    const _Float16* fa = featA + (size_t)c * 2048 + lane * 16;
    #pragma unroll
    for (int m = 0; m < 4; ++m) {
      v16h a = *(const v16h*)(fa + m * 512);
      acc[m][0] = __builtin_amdgcn_wmma_f32_16x16x32_f16(
          false, a, false, b0, (short)0, acc[m][0], false, false);
      acc[m][1] = __builtin_amdgcn_wmma_f32_16x16x32_f16(
          false, a, false, b1, (short)0, acc[m][1], false, false);
    }
  }
  // C/D layout: vgpr r -> channel 16m + 8*hi + r, pixel p0 + 16n + lo
  #pragma unroll
  for (int n = 0; n < 2; ++n) {
    const int q = p0 + n * 16 + lo;
    if (q < HW) {
      #pragma unroll
      for (int m = 0; m < 4; ++m) {
        v8h hv;
        #pragma unroll
        for (int r = 0; r < 8; ++r) hv[r] = (_Float16)acc[m][n][r];
        *(v8h*)&out[(size_t)q * 64 + m * 16 + hi * 8] = hv;
      }
    }
  }
}

// ---------------------------------------------------------------------------
// Implicit-GEMM 3x3 conv, 64->64 channels, channel-last f16 activations.
// EPI: 0 = store, 1 = relu, 2 = +res, 3 = +res then relu. res may alias out.
// ---------------------------------------------------------------------------
template <int HH, int EPI>
__global__ __launch_bounds__(256) void glss_conv(
    const _Float16* __restrict__ in, const _Float16* __restrict__ wfrag,
    const _Float16* res, _Float16* out) {
  constexpr int W = HH, HW = HH * HH;
  const int lane = threadIdx.x & 31;
  const int wid = threadIdx.x >> 5;
  const int p0 = (blockIdx.x * 8 + wid) * 32;
  if (p0 >= HW) return;
  const int hi = lane >> 4, lo = lane & 15;
  const int q0 = p0 + lo, q1 = p0 + 16 + lo;
  const int y0 = q0 / W, x0 = q0 % W;
  const int y1 = q1 / W, x1 = q1 % W;

  v8f zf = {};
  v16h z16 = {};
  v8f acc[4][2];
  #pragma unroll
  for (int m = 0; m < 4; ++m) { acc[m][0] = zf; acc[m][1] = zf; }

  for (int t = 0; t < 9; ++t) {
    const int dy = t / 3 - 1, dx = t % 3 - 1;
    const int yy0 = y0 + dy, xx0 = x0 + dx;
    const int yy1 = y1 + dy, xx1 = x1 + dx;
    const bool v0 = ((unsigned)yy0 < (unsigned)HH) && ((unsigned)xx0 < (unsigned)W) && (q0 < HW);
    const bool v1 = ((unsigned)yy1 < (unsigned)HH) && ((unsigned)xx1 < (unsigned)W) && (q1 < HW);
    #pragma unroll
    for (int kc = 0; kc < 2; ++kc) {
      v16h b0 = v0 ? *(const v16h*)&in[(size_t)(yy0 * W + xx0) * 64 + kc * 32 + hi * 16] : z16;
      v16h b1 = v1 ? *(const v16h*)&in[(size_t)(yy1 * W + xx1) * 64 + kc * 32 + hi * 16] : z16;
      const _Float16* wf = wfrag + (size_t)(t * 2 + kc) * 2048 + lane * 16;
      __builtin_prefetch(wf + 2048, 0, 1);  // next tap's fragments
      #pragma unroll
      for (int m = 0; m < 4; ++m) {
        v16h a = *(const v16h*)(wf + m * 512);
        acc[m][0] = __builtin_amdgcn_wmma_f32_16x16x32_f16(
            false, a, false, b0, (short)0, acc[m][0], false, false);
        acc[m][1] = __builtin_amdgcn_wmma_f32_16x16x32_f16(
            false, a, false, b1, (short)0, acc[m][1], false, false);
      }
    }
  }
  #pragma unroll
  for (int n = 0; n < 2; ++n) {
    const int q = p0 + n * 16 + lo;
    if (q < HW) {
      #pragma unroll
      for (int m = 0; m < 4; ++m) {
        const size_t base = (size_t)q * 64 + m * 16 + hi * 8;
        v8h rv = {};
        if constexpr (EPI >= 2) rv = *(const v8h*)&res[base];
        v8h hv;
        #pragma unroll
        for (int r = 0; r < 8; ++r) {
          float f = acc[m][n][r];
          if constexpr (EPI >= 2) f += (float)rv[r];
          if constexpr (EPI == 1 || EPI == 3) f = fmaxf(f, 0.0f);
          hv[r] = (_Float16)f;
        }
        *(v8h*)&out[base] = hv;
      }
    }
  }
}

// bilinear 2x upsample 100x100 -> 200x200, channel-last f16, half-pixel centers
__global__ __launch_bounds__(256) void glss_upsample(
    const _Float16* __restrict__ in, _Float16* __restrict__ out) {
  int idx = blockIdx.x * 256 + threadIdx.x;  // 200*200*64
  if (idx >= 200 * 200 * 64) return;
  int c = idx & 63;
  int p = idx >> 6;
  int X = p % 200, Y = p / 200;
  float fy = fminf(fmaxf((Y + 0.5f) * 0.5f - 0.5f, 0.f), 99.f);
  float fx = fminf(fmaxf((X + 0.5f) * 0.5f - 0.5f, 0.f), 99.f);
  int iy = (int)fy, ix = (int)fx;
  int iy1 = min(iy + 1, 99), ix1 = min(ix + 1, 99);
  float wy = fy - iy, wx = fx - ix;
  float v00 = (float)in[(size_t)(iy * 100 + ix) * 64 + c];
  float v01 = (float)in[(size_t)(iy * 100 + ix1) * 64 + c];
  float v10 = (float)in[(size_t)(iy1 * 100 + ix) * 64 + c];
  float v11 = (float)in[(size_t)(iy1 * 100 + ix1) * 64 + c];
  float v = (1.f - wy) * ((1.f - wx) * v00 + wx * v01) +
            wy * ((1.f - wx) * v10 + wx * v11);
  out[idx] = (_Float16)v;
}

// instance-norm stats: one block per channel over 200*200 pixels
__global__ __launch_bounds__(256) void glss_in_stats(
    const _Float16* __restrict__ v, float* __restrict__ mu,
    float* __restrict__ inv) {
  int c = blockIdx.x;
  int tid = threadIdx.x;
  float s = 0.f, ss = 0.f;
  for (int p = tid; p < 40000; p += 256) {
    float x = (float)v[(size_t)p * 64 + c];
    s += x;
    ss += x * x;
  }
  __shared__ float sh[256], sh2[256];
  sh[tid] = s; sh2[tid] = ss;
  __syncthreads();
  for (int o = 128; o > 0; o >>= 1) {
    if (tid < o) { sh[tid] += sh[tid + o]; sh2[tid] += sh2[tid + o]; }
    __syncthreads();
  }
  if (tid == 0) {
    float m = sh[0] / 40000.f;
    float var = sh2[0] / 40000.f - m * m;
    mu[c] = m;
    inv[c] = rsqrtf(var + 1e-5f);
  }
}

__global__ __launch_bounds__(256) void glss_in_add(
    const _Float16* __restrict__ v, const float* __restrict__ mu,
    const float* __restrict__ inv, const _Float16* __restrict__ skip,
    _Float16* __restrict__ out) {
  int idx = blockIdx.x * 256 + threadIdx.x;
  if (idx >= 40000 * 64) return;
  int c = idx & 63;
  out[idx] = (_Float16)(((float)v[idx] - mu[c]) * inv[c] + (float)skip[idx]);
}

// channel-last f16 -> NCHW f32 output
__global__ __launch_bounds__(256) void glss_finalize(
    const _Float16* __restrict__ x, float* __restrict__ out) {
  int idx = blockIdx.x * 256 + threadIdx.x;  // 64*40000
  if (idx >= 64 * 40000) return;
  int c = idx / 40000;
  int p = idx - c * 40000;
  out[idx] = (float)x[(size_t)p * 64 + c];
}

__global__ __launch_bounds__(256) void glss_numg(
    const float* __restrict__ means, const float* __restrict__ opac,
    float* __restrict__ out) {
  int tid = threadIdx.x;
  float s = 0.f;
  for (int g = tid; g < G_TOT; g += 256) {
    float mx = means[g * 3 + 0], my = means[g * 3 + 1], mz = means[g * 3 + 2];
    bool pos = (mx >= -50.f) && (mx <= 50.f) && (my >= -50.f) && (my <= 50.f) &&
               (mz >= -4.f) && (mz <= 4.f);
    if (pos) {
      if (opac[g * 2 + 0] > 0.05f) s += 1.f;
      if (opac[g * 2 + 1] > 0.05f) s += 1.f;
    }
  }
  __shared__ float sh[256];
  sh[tid] = s;
  __syncthreads();
  for (int o = 128; o > 0; o >>= 1) {
    if (tid < o) sh[tid] += sh[tid + o];
    __syncthreads();
  }
  if (tid == 0) out[0] = sh[0];  // B == 1 -> mean == sum
}

// ---------------------------------------------------------------------------
extern "C" void kernel_launch(void* const* d_in, const int* in_sizes, int n_in,
                              void* d_out, int out_size, void* d_ws,
                              size_t ws_size, hipStream_t stream) {
  const float* features = (const float*)d_in[0];  // (1,2048,64)
  const float* means3D  = (const float*)d_in[1];  // (1,2048,3)
  const float* cov3D    = (const float*)d_in[2];  // (1,2048,6)
  const float* opac     = (const float*)d_in[3];  // (1,2048,2)
  const float* bev_emb  = (const float*)d_in[4];  // (10000,64)
  const float* conv1_w  = (const float*)d_in[5];  // (2,64,64,3,3)
  const float* blk_w    = (const float*)d_in[6];  // (2,2,2,64,64,3,3)
  const float* up_w     = (const float*)d_in[7];  // (1,64,64,3,3)
  float* out = (float*)d_out;

  size_t off = 0;
  auto wsalloc = [&](size_t bytes) -> char* {
    char* p = (char*)d_ws + off;
    off += (bytes + 255) & ~(size_t)255;
    return p;
  };
  float*     prm0  = (float*)wsalloc(6 * G_TOT * 4);
  float*     prm1  = (float*)wsalloc(6 * G_TOT * 4);
  _Float16*  featA = (_Float16*)wsalloc(131072 * 2);
  _Float16*  wfrag = (_Float16*)wsalloc((size_t)11 * 36864 * 2);
  _Float16*  emb16 = (_Float16*)wsalloc(640000 * 2);
  _Float16*  R0    = (_Float16*)wsalloc(640000 * 2);
  _Float16*  R1    = (_Float16*)wsalloc(2560000 * 2);
  _Float16*  X0a   = (_Float16*)wsalloc(640000 * 2);
  _Float16*  X0b   = (_Float16*)wsalloc(640000 * 2);
  _Float16*  U1    = (_Float16*)wsalloc(2560000 * 2);
  _Float16*  V1    = (_Float16*)wsalloc(2560000 * 2);
  _Float16*  K1    = (_Float16*)wsalloc(2560000 * 2);
  _Float16*  X1a   = (_Float16*)wsalloc(2560000 * 2);
  _Float16*  X1b   = (_Float16*)wsalloc(2560000 * 2);
  float*     mu    = (float*)wsalloc(64 * 4);
  float*     invs  = (float*)wsalloc(64 * 4);
  (void)ws_size; (void)in_sizes; (void)n_in; (void)out_size;

  auto wf = [&](int i) { return wfrag + (size_t)i * 36864; };

  // ---- prep
  glss_prep_params<<<8, 256, 0, stream>>>(means3D, cov3D, opac, prm0, prm1);
  glss_prep_featA<<<512, 256, 0, stream>>>(features, featA);
  glss_prep_wfrag<<<1584, 256, 0, stream>>>(conv1_w, blk_w, up_w, wfrag);
  glss_prep_emb<<<2500, 256, 0, stream>>>(bev_emb, emb16);

  // ---- splatting renders (channel-last f16)
  glss_render<100><<<40, 256, 0, stream>>>(prm0, featA, R0);    // 10000 px
  glss_render<200><<<157, 256, 0, stream>>>(prm1, featA, R1);   // 40000 px

  // ---- stage 0 @ 100x100: x = emb + conv(bev0); 2 basic blocks
  glss_conv<100, 2><<<40, 256, 0, stream>>>(R0, wf(0), emb16, X0a);
  glss_conv<100, 1><<<40, 256, 0, stream>>>(X0a, wf(2), nullptr, X0b);
  glss_conv<100, 3><<<40, 256, 0, stream>>>(X0b, wf(3), X0a, X0a);
  glss_conv<100, 1><<<40, 256, 0, stream>>>(X0a, wf(4), nullptr, X0b);
  glss_conv<100, 3><<<40, 256, 0, stream>>>(X0b, wf(5), X0a, X0a);

  // ---- stage 1 @ 200x200: x = IN(conv(up(x))) + conv(bev1); 2 basic blocks
  glss_conv<200, 0><<<157, 256, 0, stream>>>(R1, wf(1), nullptr, K1);
  glss_upsample<<<10000, 256, 0, stream>>>(X0a, U1);
  glss_conv<200, 0><<<157, 256, 0, stream>>>(U1, wf(10), nullptr, V1);
  glss_in_stats<<<64, 256, 0, stream>>>(V1, mu, invs);
  glss_in_add<<<10000, 256, 0, stream>>>(V1, mu, invs, K1, X1a);
  glss_conv<200, 1><<<157, 256, 0, stream>>>(X1a, wf(6), nullptr, X1b);
  glss_conv<200, 3><<<157, 256, 0, stream>>>(X1b, wf(7), X1a, X1a);
  glss_conv<200, 1><<<157, 256, 0, stream>>>(X1a, wf(8), nullptr, X1b);
  glss_conv<200, 3><<<157, 256, 0, stream>>>(X1b, wf(9), X1a, X1a);

  // ---- outputs
  glss_finalize<<<10000, 256, 0, stream>>>(X1a, out);
  glss_numg<<<1, 256, 0, stream>>>(means3D, opac, out + 2560000);
}
